// InfiniteMultiHeadAttention_76768245449619
// MI455X (gfx1250) — matrix operand order repaired and checked by reading, not compiled, not run on hardware
//
#include <hip/hip_runtime.h>
#include <hip/hip_bf16.h>

typedef __bf16 bf16_t;
typedef bf16_t vbf16 __attribute__((ext_vector_type(16)));
typedef bf16_t vbf8  __attribute__((ext_vector_type(8)));
typedef float  vf8   __attribute__((ext_vector_type(8)));
typedef int    v4i   __attribute__((vector_size(16)));

#define D_MODEL 1024
#define NT_ROWS 16384   /* N*T = 4*4096 */
#define T_SEQ   4096
#define SEG_S   256
#define HD      64
#define N_SEGS  16

// ---- CDNA5 async global->LDS path (guarded; falls back to VGPR bounce) ----
#if defined(__gfx1250__) && \
    __has_builtin(__builtin_amdgcn_global_load_async_to_lds_b128) && \
    __has_builtin(__builtin_amdgcn_s_wait_asynccnt)
#define USE_ASYNC_LDS 1
#define ASYNC_WAIT(n) __builtin_amdgcn_s_wait_asynccnt(n)
#else
#define USE_ASYNC_LDS 0
#define ASYNC_WAIT(n) do {} while (0)
#endif

__device__ __forceinline__ void copy16_to_lds(const void* g, void* l) {
#if USE_ASYNC_LDS
  __builtin_amdgcn_global_load_async_to_lds_b128((v4i*)g, (v4i*)l, 0, 0);
#else
  *(vbf8*)l = *(const vbf8*)g;
#endif
}

union FragCvt { vbf16 v; vbf8 h[2]; };

// A-fragment (16x32 bf16): lanes 0-15 rows 0-15 hold K 0-7 & 16-23; lanes 16-31 hold K 8-15 & 24-31
__device__ __forceinline__ vbf16 load_frag_a(const bf16_t* p, int ld) {
  const int lane = threadIdx.x & 31;
  const int m  = lane & 15;
  const int kb = (lane >> 4) << 3;            // 0 or 8
  const bf16_t* r = p + m * ld + kb;
  FragCvt u;
  u.h[0] = *(const vbf8*)(r);
  u.h[1] = *(const vbf8*)(r + 16);
  return u.v;
}

// B-fragment (32x16 bf16) from BT layout (row = N column, contiguous over K)
__device__ __forceinline__ vbf16 load_frag_b(const bf16_t* p, int ld) {
  const int lane = threadIdx.x & 31;
  const int nn = lane & 15;
  const int kb = (lane >> 4) << 3;            // 0 or 8
  const bf16_t* r = p + nn * ld + kb;
  FragCvt u;
  u.h[0] = *(const vbf8*)(r);
  u.h[1] = *(const vbf8*)(r + 16);
  return u.v;
}

__device__ __forceinline__ float elu1(float x) {
  return (x > 0.f ? x : __expf(x) - 1.f) + 1.f;
}

__device__ __forceinline__ vbf16 frag_elu1(vbf16 a) {
#pragma unroll
  for (int i = 0; i < 16; ++i) a[i] = (bf16_t)elu1((float)a[i]);
  return a;
}

#define WMMA_BF16(a, b, c) \
  __builtin_amdgcn_wmma_f32_16x16x32_bf16(false, (a), false, (b), (short)0, (c), false, false)

// ---------------- LayerNorm (f32 in -> normalized bf16 out) ----------------
__global__ __launch_bounds__(256) void ln_kernel(const float* __restrict__ X,
                                                 const float* __restrict__ w,
                                                 const float* __restrict__ b,
                                                 bf16_t* __restrict__ Y) {
  __shared__ float s1[256], s2[256];
  const int row = blockIdx.x, tid = threadIdx.x;
  const float* x = X + (size_t)row * D_MODEL;
  float4 xv = *(const float4*)(x + tid * 4);
  float s = xv.x + xv.y + xv.z + xv.w;
  float q = xv.x * xv.x + xv.y * xv.y + xv.z * xv.z + xv.w * xv.w;
  s1[tid] = s; s2[tid] = q;
  __syncthreads();
  for (int off = 128; off > 0; off >>= 1) {
    if (tid < off) { s1[tid] += s1[tid + off]; s2[tid] += s2[tid + off]; }
    __syncthreads();
  }
  const float mu  = s1[0] * (1.f / D_MODEL);
  const float var = s2[0] * (1.f / D_MODEL) - mu * mu;
  const float rs  = rsqrtf(var + 1e-5f);
  bf16_t* y = Y + (size_t)row * D_MODEL + tid * 4;
  const float xs[4] = {xv.x, xv.y, xv.z, xv.w};
#pragma unroll
  for (int i = 0; i < 4; ++i)
    y[i] = (bf16_t)((xs[i] - mu) * rs * w[tid * 4 + i] + b[tid * 4 + i]);
}

// ---------------- Weight convert + transpose (f32 W -> bf16 W^T) ----------------
__global__ __launch_bounds__(256) void wconv_kernel(const float* __restrict__ W,
                                                    bf16_t* __restrict__ WT) {
  const int idx = blockIdx.x * 256 + threadIdx.x;  // 1M elements
  const int r = idx >> 10, c = idx & 1023;
  WT[(size_t)c * D_MODEL + r] = (bf16_t)W[idx];
}

// ============ GEMM core: 128x64 tile, A register-pipelined, B double-buffered in LDS ============
// Bbuf tile: 64 cols x 32 K  (ld = 32)
template <int WRITE_F32>
__device__ __forceinline__ void gemm_core(const bf16_t* __restrict__ A,
                                          const bf16_t* __restrict__ BT,
                                          bf16_t* __restrict__ Cb,
                                          float* __restrict__ Cf,
                                          int colblk) {
  __shared__ bf16_t Bbuf[2][64 * 32];
  const int tid = threadIdx.x, wave = tid >> 5, lane = tid & 31;
  const int lm = lane & 15, lo8 = (lane >> 4) * 8;
  const int r0 = blockIdx.x * 128 + wave * 16;
  const bf16_t* Ab = A + (size_t)r0 * D_MODEL;

  // cooperative B copy: each thread owns 8 contiguous bf16 (16B)
  const int bc = tid >> 2, bx = (tid & 3) * 8;
  const bf16_t* Bsrc = BT + (size_t)(colblk * 64 + bc) * D_MODEL + bx;
  bf16_t* Bd[2] = { &Bbuf[0][bc * 32 + bx], &Bbuf[1][bc * 32 + bx] };

  copy16_to_lds(Bsrc, Bd[0]);                     // tile 0 in flight
  vbf16 a_cur = load_frag_a(Ab, D_MODEL);
  vf8 acc[4] = {};

  for (int kk = 0; kk < 32; ++kk) {
    if (kk + 1 < 32) {
      copy16_to_lds(Bsrc + (kk + 1) * 32, Bd[(kk + 1) & 1]);
      ASYNC_WAIT(1);                              // tile kk complete; kk+1 in flight
    } else {
      ASYNC_WAIT(0);
    }
    __syncthreads();                              // tile kk visible to every wave
    vbf16 a_nxt = a_cur;
    if (kk + 1 < 32) a_nxt = load_frag_a(Ab + (kk + 1) * 32, D_MODEL);
    const bf16_t* Bt = Bbuf[kk & 1];
#pragma unroll
    for (int jn = 0; jn < 4; ++jn) {
      vbf16 b = load_frag_b(Bt + jn * 16 * 32, 32);
      acc[jn] = WMMA_BF16(a_cur, b, acc[jn]);
    }
    a_cur = a_nxt;
    __syncthreads();                              // reads done before buffer is overwritten
  }

#pragma unroll
  for (int jn = 0; jn < 4; ++jn)
#pragma unroll
    for (int j = 0; j < 8; ++j) {
      const int row = r0 + lo8 + j;
      if (WRITE_F32) {
        Cf[(size_t)row * D_MODEL + colblk * 64 + jn * 16 + lm] = acc[jn][j];
      } else {
        // permuted bf16 store into [N,H,T,64] head-major layout
        const int nb = row >> 12, t = row & 4095;
        const int e = jn * 16 + lm;
        Cb[((size_t)(nb * 16 + colblk) * T_SEQ + t) * HD + e] = (bf16_t)acc[jn][j];
      }
    }
}

__global__ __launch_bounds__(256) void gemm_proj(const bf16_t* __restrict__ A,
                                                 const bf16_t* __restrict__ BT,
                                                 bf16_t* __restrict__ C) {
  gemm_core<0>(A, BT, C, nullptr, blockIdx.y);
}

__global__ __launch_bounds__(256) void gemm_out(const bf16_t* __restrict__ A,
                                                const bf16_t* __restrict__ BT,
                                                float* __restrict__ C) {
  gemm_core<1>(A, BT, nullptr, C, blockIdx.y);
}

// ---------------- Segmented Infini-attention: one block per (n,h) ----------------
__global__ __launch_bounds__(256) void attn_kernel(const bf16_t* __restrict__ Q,
                                                   const bf16_t* __restrict__ K,
                                                   const bf16_t* __restrict__ V,
                                                   const float* __restrict__ beta,
                                                   bf16_t* __restrict__ O) {
  const int nh = blockIdx.x, n = nh >> 4, h = nh & 15;
  const int tid = threadIdx.x, wave = tid >> 5, lane = tid & 31;
  const int lm = lane & 15, lo8 = (lane >> 4) * 8;

  const size_t plane = (size_t)nh * T_SEQ * HD;
  const bf16_t* Qp = Q + plane;
  const bf16_t* Kp = K + plane;
  const bf16_t* Vp = V + plane;

  __shared__ bf16_t sh_q[SEG_S * HD];     // 32K  q rows (A frags for QK^T & retrieval)
  __shared__ bf16_t sh_k[SEG_S * HD];     // 32K  k rows (B frags for QK^T)
  __shared__ bf16_t sh_vT[HD * SEG_S];    // 32K  v^T (B frags for PV & mem update)
  __shared__ float  sh_mem[HD * HD];      // 16K  persistent compressive memory (d,e)
  __shared__ bf16_t sh_memT[HD * HD];     // 8K   mem^T in bf16 (B frags for retrieval)
  __shared__ float  sh_z[HD];
  __shared__ float  sh_den[SEG_S];
  __shared__ union {
    struct {
      float  scores[16 * SEG_S];          // 16K per-strip raw scores
      bf16_t probs[16 * SEG_S];           // 8K  unnormalized exp() probs
      float  attn[16 * HD];               // 4K
      float  num[16 * HD];                // 4K
      float  rsum[16];
      float  red[16 * 16];
    } s;
    bf16_t skT[HD * SEG_S];               // 32K sk^T, built after strips (A frags for mem update)
  } u;

  const float gate = 1.f / (1.f + __expf(-beta[0]));

  for (int i = tid; i < HD * HD; i += 256) sh_mem[i] = 0.f;
  if (tid < HD) sh_z[tid] = 0.f;
  __syncthreads();

  for (int seg = 0; seg < N_SEGS; ++seg) {
    const int t0 = seg * SEG_S;

    // ---- stage q,k via async bulk copy (identity layout, 32KB each) ----
    {
      const char* gq = (const char*)(Qp + (size_t)t0 * HD);
      const char* gk = (const char*)(Kp + (size_t)t0 * HD);
      char* lq = (char*)sh_q;
      char* lk = (char*)sh_k;
#pragma unroll
      for (int ch = 0; ch < 8; ++ch) {
        const int off = ch * 4096 + tid * 16;
        copy16_to_lds(gq + off, lq + off);
        copy16_to_lds(gk + off, lk + off);
      }
      // v: transpose-scatter into v^T
      const bf16_t* vr = Vp + (size_t)(t0 + tid) * HD;
#pragma unroll
      for (int d8 = 0; d8 < HD; d8 += 8) {
        vbf8 vv = *(const vbf8*)(vr + d8);
#pragma unroll
        for (int jj = 0; jj < 8; ++jj) sh_vT[(d8 + jj) * SEG_S + tid] = vv[jj];
      }
      // mem^T in bf16 for retrieval B-fragments
      for (int i = tid; i < HD * HD; i += 256) {
        const int dd = i >> 6, ee = i & 63;
        sh_memT[ee * HD + dd] = (bf16_t)sh_mem[i];
      }
      // prefetch next segment while this one computes
      if (seg + 1 < N_SEGS) {
        __builtin_prefetch(Qp + (size_t)(t0 + SEG_S + tid) * HD, 0, 0);
        __builtin_prefetch(Kp + (size_t)(t0 + SEG_S + tid) * HD, 0, 0);
        __builtin_prefetch(Vp + (size_t)(t0 + SEG_S + tid) * HD, 0, 0);
      }
      ASYNC_WAIT(0);
    }
    __syncthreads();

    // den = sq . z + eps (needs staged q)
    {
      float acc = 1e-6f;
      const bf16_t* qrow = sh_q + tid * HD;
      for (int d = 0; d < HD; ++d) acc += elu1((float)qrow[d]) * sh_z[d];
      sh_den[tid] = acc;
    }
    __syncthreads();

    // ---- 16 strips of 16 queries ----
    for (int ms = 0; ms < 16; ++ms) {
      // QK^T (causal-pruned tiles), scale 1/sqrt(64)
      for (int jn = wave; jn <= ms; jn += 8) {
        vf8 acc = {};
#pragma unroll
        for (int kk = 0; kk < 2; ++kk) {
          vbf16 a = load_frag_a(sh_q + (ms * 16) * HD + kk * 32, HD);
          vbf16 b = load_frag_b(sh_k + (jn * 16) * HD + kk * 32, HD);
          acc = WMMA_BF16(a, b, acc);
        }
#pragma unroll
        for (int j = 0; j < 8; ++j)
          u.s.scores[(lo8 + j) * SEG_S + jn * 16 + lm] = acc[j] * 0.125f;
      }
      __syncthreads();

      // softmax over valid cols (t <= s), unnormalized; 1/rowsum folded into combine
      const int r = tid >> 4, c = tid & 15;
      const int sloc = ms * 16 + r;
      float mx = -3.0e38f;
      for (int t = c; t < SEG_S; t += 16)
        if (t <= sloc) mx = fmaxf(mx, u.s.scores[r * SEG_S + t]);
      u.s.red[r * 16 + c] = mx;
      __syncthreads();
#pragma unroll
      for (int k = 0; k < 16; ++k) mx = fmaxf(mx, u.s.red[r * 16 + k]);
      __syncthreads();
      float sum = 0.f;
      for (int t = c; t < SEG_S; t += 16) {
        float p = 0.f;
        if (t <= sloc) { p = __expf(u.s.scores[r * SEG_S + t] - mx); sum += p; }
        u.s.probs[r * SEG_S + t] = (bf16_t)p;
      }
      u.s.red[r * 16 + c] = sum;
      __syncthreads();
      if (c == 0) {
        float rsm = 0.f;
#pragma unroll
        for (int k = 0; k < 16; ++k) rsm += u.s.red[r * 16 + k];
        u.s.rsum[r] = rsm;
      }
      __syncthreads();

      // PV (waves 0-3) and memory retrieval sq@mem^T (waves 4-7)
      const int nk = (16 * ms + 47) >> 5;          // ceil(16*(ms+1)/32)
      if (wave < 4) {
        const int jn = wave;
        vf8 acc = {};
        for (int kt = 0; kt < nk; ++kt) {
          vbf16 a = load_frag_a(u.s.probs + kt * 32, SEG_S);
          vbf16 b = load_frag_b(sh_vT + (jn * 16) * SEG_S + kt * 32, SEG_S);
          acc = WMMA_BF16(a, b, acc);
        }
#pragma unroll
        for (int j = 0; j < 8; ++j) u.s.attn[(lo8 + j) * HD + jn * 16 + lm] = acc[j];
      } else {
        const int jn = wave - 4;
        vf8 acc = {};
#pragma unroll
        for (int kk = 0; kk < 2; ++kk) {
          vbf16 a = frag_elu1(load_frag_a(sh_q + (ms * 16) * HD + kk * 32, HD));
          vbf16 b = load_frag_b(sh_memT + (jn * 16) * HD + kk * 32, HD);
          acc = WMMA_BF16(a, b, acc);
        }
#pragma unroll
        for (int j = 0; j < 8; ++j) u.s.num[(lo8 + j) * HD + jn * 16 + lm] = acc[j];
      }
      __syncthreads();

      // combine & write out (bf16, [N,T,H*64] layout for the output GEMM)
#pragma unroll
      for (int jj = 0; jj < 4; ++jj) {
        const int elem = jj * 256 + tid;
        const int i = elem >> 6, e = elem & 63;
        const int sl = ms * 16 + i;
        const float o = gate * u.s.num[elem] / sh_den[sl] +
                        (1.f - gate) * u.s.attn[elem] / u.s.rsum[i];
        O[((size_t)n * T_SEQ + t0 + sl) * D_MODEL + h * 64 + e] = (bf16_t)o;
      }
      __syncthreads();
    }

    // ---- sk^T into aliased region (strip buffers now dead) ----
    for (int idx = tid; idx < HD * SEG_S; idx += 256) {
      const int d = idx >> 8, t = idx & 255;
      u.skT[idx] = (bf16_t)elu1((float)sh_k[t * HD + d]);
    }
    __syncthreads();

    // ---- mem += sk^T @ v ; z += sum_s sk ----
    for (int tile = wave; tile < 16; tile += 8) {
      const int jd = tile >> 2, je = tile & 3;
      vf8 acc = {};
#pragma unroll
      for (int kt = 0; kt < 8; ++kt) {
        vbf16 a = load_frag_a(u.skT + (jd * 16) * SEG_S + kt * 32, SEG_S);
        vbf16 b = load_frag_b(sh_vT + (je * 16) * SEG_S + kt * 32, SEG_S);
        acc = WMMA_BF16(a, b, acc);
      }
#pragma unroll
      for (int j = 0; j < 8; ++j)
        sh_mem[(jd * 16 + lo8 + j) * HD + je * 16 + lm] += acc[j];
    }
    if (tid < HD) {
      float zs = 0.f;
      for (int t = 0; t < SEG_S; ++t) zs += (float)u.skT[tid * SEG_S + t];
      sh_z[tid] += zs;
    }
    __syncthreads();
  }
}

extern "C" void kernel_launch(void* const* d_in, const int* in_sizes, int n_in,
                              void* d_out, int out_size, void* d_ws, size_t ws_size,
                              hipStream_t stream) {
  (void)in_sizes; (void)n_in; (void)out_size; (void)ws_size;
  const float* q_in  = (const float*)d_in[0];
  const float* k_in  = (const float*)d_in[1];
  const float* v_in  = (const float*)d_in[2];
  const float* Wq    = (const float*)d_in[3];
  const float* Wk    = (const float*)d_in[4];
  const float* Wv    = (const float*)d_in[5];
  const float* Wo    = (const float*)d_in[6];
  const float* ln_w  = (const float*)d_in[7];
  const float* ln_b  = (const float*)d_in[8];
  const float* beta  = (const float*)d_in[9];
  float* out = (float*)d_out;

  const size_t big = (size_t)NT_ROWS * D_MODEL;   // 16,777,216 bf16 elems
  bf16_t* bufA = (bf16_t*)d_ws;                   // reusable activation buffer
  bf16_t* bufQ = bufA + big;
  bf16_t* bufK = bufQ + big;
  bf16_t* bufV = bufK + big;
  bf16_t* WqT  = bufV + big;
  bf16_t* WkT  = WqT + (size_t)(1 << 20);
  bf16_t* WvT  = WkT + (size_t)(1 << 20);
  bf16_t* WoT  = WvT + (size_t)(1 << 20);

  const dim3 blk(256);
  // weights -> bf16 transposed
  wconv_kernel<<<4096, blk, 0, stream>>>(Wq, WqT);
  wconv_kernel<<<4096, blk, 0, stream>>>(Wk, WkT);
  wconv_kernel<<<4096, blk, 0, stream>>>(Wv, WvT);
  wconv_kernel<<<4096, blk, 0, stream>>>(Wo, WoT);

  const dim3 gGemm(128, 16);
  // q path
  ln_kernel<<<NT_ROWS, blk, 0, stream>>>(q_in, ln_w, ln_b, bufA);
  gemm_proj<<<gGemm, blk, 0, stream>>>(bufA, WqT, bufQ);
  // k path
  ln_kernel<<<NT_ROWS, blk, 0, stream>>>(k_in, ln_w, ln_b, bufA);
  gemm_proj<<<gGemm, blk, 0, stream>>>(bufA, WkT, bufK);
  // v path
  ln_kernel<<<NT_ROWS, blk, 0, stream>>>(v_in, ln_w, ln_b, bufA);
  gemm_proj<<<gGemm, blk, 0, stream>>>(bufA, WvT, bufV);

  // segmented infini-attention (one block per (n,h))
  attn_kernel<<<64, blk, 0, stream>>>(bufQ, bufK, bufV, beta, bufA);

  // output projection (f32 result)
  gemm_out<<<gGemm, blk, 0, stream>>>(bufA, WoT, out);
}
